// LSTMAttentionCell_43112881717814
// MI455X (gfx1250) — matrix operand order repaired and checked by reading, not compiled, not run on hardware
//
#include <hip/hip_runtime.h>
#include <hip/hip_bf16.h>

#define BB 4096
#define CHAR_LEN 64
#define LSTM_N 400
#define VOCAB 73
#define NATTN 10

typedef __attribute__((ext_vector_type(16))) __bf16 v16bf;
typedef __attribute__((ext_vector_type(8)))  float  v8f;
typedef __attribute__((ext_vector_type(8)))  unsigned int v8u;

__device__ __forceinline__ unsigned short f2bf(float f) {
    unsigned int u = __builtin_bit_cast(unsigned int, f);
    unsigned int r = u + 0x7FFFu + ((u >> 16) & 1u);
    return (unsigned short)(r >> 16);
}

// ---- WMMA fragment loads -------------------------------------------------
// A (16x32 bf16), per ISA 7.12.2: lane L: m=L%16, h=L/16; VGPR p (pairs):
//   k = (p<4 ? 2p : 16+2(p-4)) + 8h
__device__ __forceinline__ v16bf load_a_frag_lds(const unsigned short* Ash,
                                                 int Kpad, int k0) {
    int lane = threadIdx.x & 31;
    int m = lane & 15, h = lane >> 4;
    const unsigned short* row = Ash + m * Kpad;
    v8u u;
#pragma unroll
    for (int p = 0; p < 8; ++p) {
        int k = (p < 4 ? 2 * p : 16 + 2 * (p - 4)) + 8 * h + k0;
        u[p] = *(const unsigned int*)(row + k);   // pair of bf16
    }
    return __builtin_bit_cast(v16bf, u);
}

__device__ __forceinline__ v16bf load_a_frag_glb(const unsigned short* X,
                                                 int Kpad, int row0, int k0) {
    int lane = threadIdx.x & 31;
    int m = lane & 15, h = lane >> 4;
    const unsigned short* row = X + (size_t)(row0 + m) * Kpad;
    v8u u;
#pragma unroll
    for (int p = 0; p < 8; ++p) {
        int k = (p < 4 ? 2 * p : 16 + 2 * (p - 4)) + 8 * h + k0;
        u[p] = *(const unsigned int*)(row + k);
    }
    return __builtin_bit_cast(v16bf, u);
}

// B (32x16 bf16): lane L: n=L%16, h=L/16; holds K = 8h..8h+7, 16+8h..16+8h+7
// from weight row n (weights stored [N,K] row-major, since out = x * W^T).
__device__ __forceinline__ v16bf load_b_frag(const unsigned short* W,
                                             int Kpad, int nrow, int k0) {
    int lane = threadIdx.x & 31;
    int n = lane & 15, h = lane >> 4;
    const unsigned short* p = W + (size_t)(nrow + n) * Kpad + k0 + 8 * h;
    uint4 lo = *(const uint4*)p;
    uint4 hi = *(const uint4*)(p + 16);
    v8u u;
    u[0] = lo.x; u[1] = lo.y; u[2] = lo.z; u[3] = lo.w;
    u[4] = hi.x; u[5] = hi.y; u[6] = hi.z; u[7] = hi.w;
    return __builtin_bit_cast(v16bf, u);
}

// ---- CDNA5 async global->LDS copy (ASYNCcnt path, bypasses VGPRs) --------
__device__ __forceinline__ void async_b128_to_lds(unsigned int lds_byte_addr,
                                                  const void* gaddr) {
    // LDS dest address comes from the vdst VGPR; flat shared addr low 32 bits = LDS offset
    asm volatile("global_load_async_to_lds_b128 %0, %1, off"
                 :: "v"(lds_byte_addr), "v"(gaddr) : "memory");
}
__device__ __forceinline__ void wait_asynccnt0() {
    asm volatile("s_wait_asynccnt 0x0" ::: "memory");
}

// ---- Prep: weights fp32 -> bf16, [W_ih | W_hh | pad] fused ----------------
__global__ void build_weights_kernel(unsigned short* __restrict__ dst,
                                     const float* __restrict__ A, int KA,
                                     const float* __restrict__ Bm, int KB,
                                     int N, int Nsrc, int Kpad) {
    size_t total = (size_t)N * Kpad;
    for (size_t i = (size_t)blockIdx.x * blockDim.x + threadIdx.x; i < total;
         i += (size_t)gridDim.x * blockDim.x) {
        int n = (int)(i / Kpad), k = (int)(i % Kpad);
        float v = 0.f;
        if (n < Nsrc) {
            if (k < KA)            v = A[(size_t)n * KA + k];
            else if (k < KA + KB)  v = Bm[(size_t)n * KB + (k - KA)];
        }
        dst[i] = f2bf(v);
    }
}

// ---- Prep: activation matrix [s0 | s1 | s2 | zero-pad], bf16 --------------
__global__ void build_x_kernel(unsigned short* __restrict__ dst, int Kpad,
                               const float* __restrict__ s0, int w0,
                               const float* __restrict__ s1, int w1,
                               const float* __restrict__ s2, int w2) {
    size_t total = (size_t)BB * Kpad;
    for (size_t i = (size_t)blockIdx.x * blockDim.x + threadIdx.x; i < total;
         i += (size_t)gridDim.x * blockDim.x) {
        int b = (int)(i / Kpad), k = (int)(i % Kpad);
        float v = 0.f;
        if (k < w0)                { if (s0) v = s0[(size_t)b * w0 + k]; }
        else if (k < w0 + w1)      { if (s1) v = s1[(size_t)b * w1 + (k - w0)]; }
        else if (k < w0 + w1 + w2) { if (s2) v = s2[(size_t)b * w2 + (k - w0 - w1)]; }
        dst[i] = f2bf(v);
    }
}

// ---- Fused LSTM stage: GEMM (bf16 WMMA) + gates + cell update -------------
// Block: 128 threads (4 waves = 4 gates). Tile: 32 batch rows x 80 h-cols.
// Each k-step loads 5 B fragments once and reuses them for 2 A sub-tiles
// (halves L2 weight traffic vs M=16).
__global__ __launch_bounds__(128) void lstm_stage_kernel(
    const unsigned short* __restrict__ X, int Kpad,
    const unsigned short* __restrict__ Wc,      // [1600, Kpad] bf16
    const float* __restrict__ bias,             // [1600]
    const float* __restrict__ c_in,             // [B, 400]
    unsigned short* __restrict__ dst1, int stride1, int off1,
    unsigned short* __restrict__ dst2, int stride2, int off2,
    float* __restrict__ out_f32) {
    __shared__ __align__(16) unsigned short Ash[32 * 896];
    __shared__ float Gsh[4][32 * 80];

    const int b0  = blockIdx.x * 32;
    const int nh0 = blockIdx.y * 80;
    const int tid = threadIdx.x;
    const int wave = tid >> 5;                  // gate id 0..3

    // stage A tile (32 rows, contiguous span) into LDS via async copy
    {
        const char* src = (const char*)(X + (size_t)b0 * Kpad);
        unsigned int lds0 = (unsigned int)(size_t)(void*)Ash;
        int nchunks = (32 * Kpad) >> 3;         // 16B chunks
        for (int i = tid; i < nchunks; i += 128)
            async_b128_to_lds(lds0 + i * 16, src + (size_t)i * 16);
        wait_asynccnt0();
    }
    __syncthreads();

    v8f acc[2][5] = {};
    const int grow = wave * LSTM_N + nh0;
    for (int k0 = 0; k0 < Kpad; k0 += 32) {
        v16bf bfr[5];
#pragma unroll
        for (int t = 0; t < 5; ++t)
            bfr[t] = load_b_frag(Wc, Kpad, grow + t * 16, k0);
#pragma unroll
        for (int mi = 0; mi < 2; ++mi) {
            v16bf a = load_a_frag_lds(Ash + mi * 16 * Kpad, Kpad, k0);
#pragma unroll
            for (int t = 0; t < 5; ++t)
                acc[mi][t] = __builtin_amdgcn_wmma_f32_16x16x32_bf16(
                    false, a, false, bfr[t], (short)0, acc[mi][t], false, false);
        }
    }

    // spill this wave's gate slab: D layout lane L: n=L%16, m = j + 8*(L/16)
    {
        int lane = tid & 31, n = lane & 15, h = lane >> 4;
#pragma unroll
        for (int mi = 0; mi < 2; ++mi)
#pragma unroll
            for (int t = 0; t < 5; ++t)
#pragma unroll
                for (int j = 0; j < 8; ++j)
                    Gsh[wave][(mi * 16 + j + 8 * h) * 80 + t * 16 + n] =
                        acc[mi][t][j];
    }
    __syncthreads();

    // elementwise gates + cell update: 32*80 = 2560 elems, 20 per thread
    for (int i = tid; i < 32 * 80; i += 128) {
        int m = i / 80, n = i % 80;
        int row = b0 + m, col = nh0 + n;
        float gi = Gsh[0][i] + bias[col];
        float gf = Gsh[1][i] + bias[LSTM_N + col];
        float gg = Gsh[2][i] + bias[2 * LSTM_N + col];
        float go = Gsh[3][i] + bias[3 * LSTM_N + col];
        float ii = 1.f / (1.f + __expf(-gi));
        float ff = 1.f / (1.f + __expf(-gf));
        float g  = tanhf(gg);
        float oo = 1.f / (1.f + __expf(-go));
        float c  = c_in[(size_t)row * LSTM_N + col];
        float cn = ff * c + ii * g;
        float hn = oo * tanhf(cn);
        if (out_f32) out_f32[(size_t)row * LSTM_N + col] = hn;
        unsigned short hb = f2bf(hn);
        if (dst1) dst1[(size_t)row * stride1 + off1 + col] = hb;
        if (dst2) dst2[(size_t)row * stride2 + off2 + col] = hb;
    }
}

// ---- Attention params: [B,480]x[480,32] WMMA + softplus/clamp/kappa -------
__global__ __launch_bounds__(32) void attn_params_kernel(
    const unsigned short* __restrict__ XA,   // [B,480] bf16
    const unsigned short* __restrict__ Wac,  // [32,480] bf16
    const float* __restrict__ b_attn,        // [30]
    const float* __restrict__ kappa_in,      // [B,10]
    float* __restrict__ params) {            // [B,32]
    const int b0 = blockIdx.x * 16;
    v8f acc[2] = {};
    for (int k0 = 0; k0 < 480; k0 += 32) {
        v16bf a = load_a_frag_glb(XA, 480, b0, k0);
#pragma unroll
        for (int t = 0; t < 2; ++t) {
            v16bf b = load_b_frag(Wac, 480, t * 16, k0);
            acc[t] = __builtin_amdgcn_wmma_f32_16x16x32_bf16(
                false, a, false, b, (short)0, acc[t], false, false);
        }
    }
    int lane = threadIdx.x, n = lane & 15, h = lane >> 4;
#pragma unroll
    for (int t = 0; t < 2; ++t) {
        int col = t * 16 + n;
#pragma unroll
        for (int j = 0; j < 8; ++j) {
            int row = b0 + j + 8 * h;
            float r = 0.f;
            if (col < 3 * NATTN) {
                float v = acc[t][j] + b_attn[col];
                float sp = (v > 20.f) ? v : log1pf(__expf(v));
                if (col < NATTN)          r = sp;                     // alpha
                else if (col < 2 * NATTN) r = fmaxf(sp, 0.01f);       // beta
                else r = kappa_in[(size_t)row * NATTN + (col - 2 * NATTN)]
                         + sp * 0.04f;                                // kappa
            }
            params[(size_t)row * 32 + col] = r;
        }
    }
}

// ---- phi + w = phi . attention_values ; scatter bf16 into X2/X3 -----------
__global__ __launch_bounds__(256) void attn_w_kernel(
    const float* __restrict__ params,  // [B,32]
    const float* __restrict__ AV,      // [B,64,73]
    const int* __restrict__ lens,      // [B]
    unsigned short* __restrict__ X2, int strideX, int offW,
    unsigned short* __restrict__ X3) {
    __shared__ float phish[8][CHAR_LEN];
    int wave = threadIdx.x >> 5, lane = threadIdx.x & 31;
    int b = blockIdx.x * 8 + wave;
    const float* P = params + (size_t)b * 32;
    int len = lens[b];
#pragma unroll
    for (int cc = 0; cc < 2; ++cc) {
        int c = lane + 32 * cc;
        float u = (float)c, phi = 0.f;
#pragma unroll
        for (int j = 0; j < NATTN; ++j) {
            float al = P[j], be = P[NATTN + j], ka = P[2 * NATTN + j];
            float d = ka - u;
            phi += al * __expf(-d * d / be);
        }
        phish[wave][c] = (c < len) ? phi : 0.f;
    }
    __syncthreads();
    const float* av = AV + (size_t)b * CHAR_LEN * VOCAB;
    for (int v = lane; v < VOCAB; v += 32) {
        float acc = 0.f;
        for (int c = 0; c < CHAR_LEN; ++c)
            acc += phish[wave][c] * av[c * VOCAB + v];   // coalesced over v
        unsigned short wb = f2bf(acc);
        X2[(size_t)b * strideX + offW + v] = wb;
        X3[(size_t)b * strideX + offW + v] = wb;
    }
}

// ---------------------------------------------------------------------------
static inline size_t alignup(size_t x) { return (x + 255) & ~(size_t)255; }

extern "C" void kernel_launch(void* const* d_in, const int* in_sizes, int n_in,
                              void* d_out, int out_size, void* d_ws, size_t ws_size,
                              hipStream_t stream) {
    const float* inputs = (const float*)d_in[0];
    const float* h1 = (const float*)d_in[1];
    const float* c1 = (const float*)d_in[2];
    const float* h2 = (const float*)d_in[3];
    const float* c2 = (const float*)d_in[4];
    const float* h3 = (const float*)d_in[5];
    const float* c3 = (const float*)d_in[6];
    const float* kappa = (const float*)d_in[7];
    const float* w_prev = (const float*)d_in[8];
    const float* AV = (const float*)d_in[9];
    const int*   lens = (const int*)d_in[10];
    const float* W_ih1 = (const float*)d_in[11];
    const float* W_hh1 = (const float*)d_in[12];
    const float* b1 = (const float*)d_in[13];
    const float* W_attn = (const float*)d_in[14];
    const float* b_attn = (const float*)d_in[15];
    const float* W_ih2 = (const float*)d_in[16];
    const float* W_hh2 = (const float*)d_in[17];
    const float* b2 = (const float*)d_in[18];
    const float* W_ih3 = (const float*)d_in[19];
    const float* W_hh3 = (const float*)d_in[20];
    const float* b3 = (const float*)d_in[21];

    const int K1 = 480;   // 73+3+400 = 476 -> 480
    const int K2 = 896;   // 3+400+73+400 = 876 -> 896
    const int NG = 4 * LSTM_N;  // 1600

    char* ws = (char*)d_ws;
    size_t off = 0;
    unsigned short* W1c = (unsigned short*)(ws + off); off = alignup(off + (size_t)NG * K1 * 2);
    unsigned short* W2c = (unsigned short*)(ws + off); off = alignup(off + (size_t)NG * K2 * 2);
    unsigned short* W3c = (unsigned short*)(ws + off); off = alignup(off + (size_t)NG * K2 * 2);
    unsigned short* Wac = (unsigned short*)(ws + off); off = alignup(off + (size_t)32 * K1 * 2);
    unsigned short* X1  = (unsigned short*)(ws + off); off = alignup(off + (size_t)BB * K1 * 2);
    unsigned short* XA  = (unsigned short*)(ws + off); off = alignup(off + (size_t)BB * K1 * 2);
    unsigned short* X2  = (unsigned short*)(ws + off); off = alignup(off + (size_t)BB * K2 * 2);
    unsigned short* X3  = (unsigned short*)(ws + off); off = alignup(off + (size_t)BB * K2 * 2);
    float* params       = (float*)(ws + off);          off = alignup(off + (size_t)BB * 32 * 4);
    if (off > ws_size) return;  // workspace too small (constant per environment)

    // --- weight prep (bf16, fused [W_ih | W_hh]) ---
    build_weights_kernel<<<2048, 256, 0, stream>>>(W1c, W_ih1, VOCAB + 3, W_hh1, LSTM_N, NG, NG, K1);
    build_weights_kernel<<<2048, 256, 0, stream>>>(W2c, W_ih2, 476, W_hh2, LSTM_N, NG, NG, K2);
    build_weights_kernel<<<2048, 256, 0, stream>>>(W3c, W_ih3, 476, W_hh3, LSTM_N, NG, NG, K2);
    build_weights_kernel<<<256, 256, 0, stream>>>(Wac, W_attn, 476, nullptr, 0, 32, 3 * NATTN, K1);

    // --- activation prep ---
    // X1 = [w_prev | inputs | h1 | pad]
    build_x_kernel<<<2048, 256, 0, stream>>>(X1, K1, w_prev, VOCAB, inputs, 3, h1, LSTM_N);
    // XA = [w_prev | inputs | (h1n later) | pad]
    build_x_kernel<<<2048, 256, 0, stream>>>(XA, K1, w_prev, VOCAB, inputs, 3, nullptr, LSTM_N);
    // X2 = [inputs | (h1n later) | (w later) | h2 | pad]
    build_x_kernel<<<2048, 256, 0, stream>>>(X2, K2, inputs, 3, nullptr, LSTM_N + VOCAB, h2, LSTM_N);
    // X3 = [inputs | (h2n later) | (w later) | h3 | pad]
    build_x_kernel<<<2048, 256, 0, stream>>>(X3, K2, inputs, 3, nullptr, LSTM_N + VOCAB, h3, LSTM_N);

    dim3 lgrid(BB / 32, LSTM_N / 80);

    // --- LSTM1: h1n -> XA[:,76:476], X2[:,3:403] ---
    lstm_stage_kernel<<<lgrid, 128, 0, stream>>>(
        X1, K1, W1c, b1, c1, XA, K1, VOCAB + 3, X2, K2, 3, nullptr);

    // --- attention params + w ---
    attn_params_kernel<<<BB / 16, 32, 0, stream>>>(XA, Wac, b_attn, kappa, params);
    attn_w_kernel<<<BB / 8, 256, 0, stream>>>(params, AV, lens, X2, K2, 3 + LSTM_N, X3);

    // --- LSTM2: h2n -> X3[:,3:403] ---
    lstm_stage_kernel<<<lgrid, 128, 0, stream>>>(
        X2, K2, W2c, b2, c2, X3, K2, 3, nullptr, 0, 0, nullptr);

    // --- LSTM3: h3n -> d_out (fp32) ---
    lstm_stage_kernel<<<lgrid, 128, 0, stream>>>(
        X3, K2, W3c, b3, c3, nullptr, 0, 0, nullptr, 0, 0, (float*)d_out);
}
